// ESALSTM_1030792151679
// MI455X (gfx1250) — matrix operand order repaired and checked by reading, not compiled, not run on hardware
//
#include <hip/hip_runtime.h>
#include <hip/hip_bf16.h>

// ---------------------------------------------------------------------------
// Types for CDNA5 WMMA
// ---------------------------------------------------------------------------
typedef __attribute__((ext_vector_type(16))) __bf16 v16bf;
typedef __attribute__((ext_vector_type(8)))  float  v8f;

union FragBF { v16bf v; uint4 u[2]; };

#if __has_builtin(__builtin_amdgcn_global_load_async_to_lds_b128) && \
    __has_builtin(__builtin_amdgcn_s_wait_asynccnt)
#define GEMM_ASYNC_LDS 1
// b128 async copy: params are pointers to 16-byte int vectors (per hipcc diag)
typedef int v4i_ __attribute__((vector_size(4 * sizeof(int))));
typedef __attribute__((address_space(1))) v4i_* as1_i4p;   // global src
typedef __attribute__((address_space(3))) v4i_* as3_i4p;   // LDS dst
#else
#define GEMM_ASYNC_LDS 0
#endif

__device__ __forceinline__ unsigned short f2bf(float f) {
    unsigned int u = __float_as_uint(f);
    unsigned int r = (u + 0x7FFFu + ((u >> 16) & 1u)) >> 16;
    return (unsigned short)r;
}

__device__ __forceinline__ float sigm(float x) { return 1.0f / (1.0f + expf(-x)); }

// ---------------------------------------------------------------------------
// f32 -> bf16 conversion (weight staging)
// ---------------------------------------------------------------------------
__global__ void cvt_f32_bf16(const float* __restrict__ src,
                             unsigned short* __restrict__ dst, int n) {
    int i = blockIdx.x * blockDim.x + threadIdx.x;
    if (i < n) dst[i] = f2bf(src[i]);
}

// ---------------------------------------------------------------------------
// Embedding: h[b][d] = sum_{k<4} x[b][k]*W[d][k] + b[d]  (K=4, trivial)
// writes f32 and bf16 copies
// ---------------------------------------------------------------------------
__global__ void embed_kernel(const float* __restrict__ x,
                             const float* __restrict__ W,
                             const float* __restrict__ b,
                             float* __restrict__ hf,
                             unsigned short* __restrict__ hb) {
    int i = blockIdx.x * blockDim.x + threadIdx.x;   // over 1024*1024
    int row = i >> 10;
    int d   = i & 1023;
    float s = b[d];
    const float* xr = x + row * 4;
    const float* wr = W + d * 4;
#pragma unroll
    for (int k = 0; k < 4; ++k) s += xr[k] * wr[k];
    hf[i] = s;
    hb[i] = f2bf(s);
}

// ---------------------------------------------------------------------------
// bf16 WMMA GEMM:  C(MxN) = A(MxK) * W(NxK)^T  (+bias, opt ReLU)
// A, W row-major bf16 (both dotted along K -> contiguous b128 loads).
// Block = 256 thr = 8 waves; wave w -> rows [bx*128 + w*16, +16), cols [by*64, +64).
// All 8 waves share the same 64 output columns, so the B tile (64x32 bf16 =
// 4 KB per k-step) is staged once per block in LDS:
//   - async path: global_load_async_to_lds_b128, double-buffered, ASYNCcnt
//   - fallback:  synchronous cooperative copy (load + ds_store)
// A fragments stay direct-from-global (wave-private rows) with prefetch.
// Writes f32 C (optional) and/or bf16 C (optional).
// M % 128 == 0, N % 64 == 0, K % 32 == 0.
// ---------------------------------------------------------------------------
__global__ __launch_bounds__(256)
void gemm_bf16_wmma(const unsigned short* __restrict__ A,
                    const unsigned short* __restrict__ Bw,
                    const float* __restrict__ bias,
                    float* __restrict__ Cf,
                    unsigned short* __restrict__ Cb,
                    int M, int N, int K, int relu) {
    __shared__ unsigned short lbs[2][64 * 32];   // double-buffered B tile (8 KB)

    const int tid  = threadIdx.x;
    const int wave = tid >> 5;
    const int lane = tid & 31;
    const int half = lane >> 4;      // 0: lanes 0-15, 1: lanes 16-31
    const int l16  = lane & 15;

    const int m0 = blockIdx.x * 128 + wave * 16;
    const int n0 = blockIdx.y * 64;

    // --- B-tile staging assignment: thread tid copies 16 bytes per k-step ---
    const int sn = tid >> 2;          // staging row 0..63  (column n0+sn of C)
    const int sc = (tid & 3) << 3;    // element offset within 32-elem k-slice
    const unsigned short* Bg = Bw + (size_t)(n0 + sn) * K + sc;
    unsigned short* Ls[2] = { &lbs[0][sn * 32 + sc], &lbs[1][sn * 32 + sc] };

    // A fragment: lane holds row m0+l16; e -> K=k0+half*16+e
    //             (32 contiguous bytes)
    const unsigned short* Ap = A + (size_t)(m0 + l16) * K;

    // B fragment (from LDS) tile t: lane holds column t*16+l16;
    //             e -> k_local = half*16 + e  (32 contiguous bytes)
    const int lb_off[4] = { (0 * 16 + l16) * 32 + half * 16,
                            (1 * 16 + l16) * 32 + half * 16,
                            (2 * 16 + l16) * 32 + half * 16,
                            (3 * 16 + l16) * 32 + half * 16 };

    v8f acc[4] = {};

#if GEMM_ASYNC_LDS
    // prologue: kick off tile 0 fill
    __builtin_amdgcn_global_load_async_to_lds_b128(
        (as1_i4p)(unsigned long long)(size_t)Bg,
        (as3_i4p)(unsigned int)(size_t)Ls[0], 0, 0);
#endif

    int buf = 0;
    for (int k0 = 0; k0 < K; k0 += 32) {
        // all waves done reading the buffer we are about to (re)fill
        __syncthreads();

#if GEMM_ASYNC_LDS
        if (k0 + 32 < K) {
            __builtin_amdgcn_global_load_async_to_lds_b128(
                (as1_i4p)(unsigned long long)(size_t)(Bg + k0 + 32),
                (as3_i4p)(unsigned int)(size_t)Ls[buf ^ 1], 0, 0);
            __builtin_amdgcn_s_wait_asynccnt(1);   // oldest fill (current buf) landed
        } else {
            __builtin_amdgcn_s_wait_asynccnt(0);
        }
#else
        {
            uint4 tmp = *(const uint4*)(Bg + k0);
            *(uint4*)Ls[buf] = tmp;
        }
#endif
        __syncthreads();                           // tile `buf` visible to all

        if (k0 + 32 < K) {
            __builtin_prefetch(Ap + k0 + 32 + half * 8, 0, 1);
        }

        FragBF fa;
        fa.u[0] = *(const uint4*)(Ap + k0 + half * 16);
        fa.u[1] = *(const uint4*)(Ap + k0 + half * 16 + 8);

        const unsigned short* lb = lbs[buf];
        FragBF fb0, fb1, fb2, fb3;
        fb0.u[0] = ((const uint4*)(lb + lb_off[0]))[0];
        fb0.u[1] = ((const uint4*)(lb + lb_off[0]))[1];
        fb1.u[0] = ((const uint4*)(lb + lb_off[1]))[0];
        fb1.u[1] = ((const uint4*)(lb + lb_off[1]))[1];
        fb2.u[0] = ((const uint4*)(lb + lb_off[2]))[0];
        fb2.u[1] = ((const uint4*)(lb + lb_off[2]))[1];
        fb3.u[0] = ((const uint4*)(lb + lb_off[3]))[0];
        fb3.u[1] = ((const uint4*)(lb + lb_off[3]))[1];

        acc[0] = __builtin_amdgcn_wmma_f32_16x16x32_bf16(false, fa.v, false, fb0.v,
                                                         (short)0, acc[0], false, false);
        acc[1] = __builtin_amdgcn_wmma_f32_16x16x32_bf16(false, fa.v, false, fb1.v,
                                                         (short)0, acc[1], false, false);
        acc[2] = __builtin_amdgcn_wmma_f32_16x16x32_bf16(false, fa.v, false, fb2.v,
                                                         (short)0, acc[2], false, false);
        acc[3] = __builtin_amdgcn_wmma_f32_16x16x32_bf16(false, fa.v, false, fb3.v,
                                                         (short)0, acc[3], false, false);
        buf ^= 1;
    }

    // D layout: VGPR r, lane -> row = m0 + half*8 + r, col = n0 + t*16 + l16
#pragma unroll
    for (int t = 0; t < 4; ++t) {
        const int col = n0 + t * 16 + l16;
        const float bv = bias ? bias[col] : 0.0f;
#pragma unroll
        for (int r = 0; r < 8; ++r) {
            float v = acc[t][r] + bv;
            if (relu) v = v > 0.0f ? v : 0.0f;
            size_t idx = (size_t)(m0 + half * 8 + r) * N + col;
            if (Cf) Cf[idx] = v;
            if (Cb) Cb[idx] = f2bf(v);
        }
    }
}

// ---------------------------------------------------------------------------
// Residual add + LayerNorm over E=1024; writes f32 h and bf16 h. 1 block/row.
// ---------------------------------------------------------------------------
__global__ __launch_bounds__(256)
void add_ln_kernel(float* __restrict__ h, const float* __restrict__ delta,
                   const float* __restrict__ w, const float* __restrict__ b,
                   unsigned short* __restrict__ hb, int E) {
    __shared__ float red1[256];
    __shared__ float red2[256];
    __shared__ float s_mean, s_inv;
    const int row = blockIdx.x;
    const int tid = threadIdx.x;
    float s = 0.0f, s2 = 0.0f;
    for (int j = tid; j < E; j += 256) {
        float z = h[(size_t)row * E + j] + delta[(size_t)row * E + j];
        s += z; s2 += z * z;
    }
    red1[tid] = s; red2[tid] = s2;
    __syncthreads();
    for (int off = 128; off > 0; off >>= 1) {
        if (tid < off) { red1[tid] += red1[tid + off]; red2[tid] += red2[tid + off]; }
        __syncthreads();
    }
    if (tid == 0) {
        float mean = red1[0] / (float)E;
        float var  = red2[0] / (float)E - mean * mean;
        s_mean = mean;
        s_inv  = rsqrtf(var + 1e-5f);
    }
    __syncthreads();
    const float mean = s_mean, inv = s_inv;
    for (int j = tid; j < E; j += 256) {
        size_t idx = (size_t)row * E + j;
        float z = h[idx] + delta[idx];
        float y = (z - mean) * inv * w[j] + b[j];
        h[idx]  = y;
        hb[idx] = f2bf(y);
    }
}

// ---------------------------------------------------------------------------
// Row sums over E=1024 (col_sums in reference). 1 block/row.
// ---------------------------------------------------------------------------
__global__ __launch_bounds__(256)
void rowsum_kernel(const float* __restrict__ h, float* __restrict__ out, int E) {
    __shared__ float red[256];
    const int row = blockIdx.x;
    const int tid = threadIdx.x;
    float s = 0.0f;
    for (int j = tid; j < E; j += 256) s += h[(size_t)row * E + j];
    red[tid] = s;
    __syncthreads();
    for (int off = 128; off > 0; off >>= 1) {
        if (tid < off) red[tid] += red[tid + off];
        __syncthreads();
    }
    if (tid == 0) out[row] = red[0];
}

// ---------------------------------------------------------------------------
// Top-10 of 1024 values, indices sorted ascending. Single block of 1024 thr.
// Iterative argmax with smaller-index tie break (matches jax top_k).
// ---------------------------------------------------------------------------
__global__ __launch_bounds__(1024)
void topk_kernel(const float* __restrict__ vals, int* __restrict__ top) {
    __shared__ float sv[1024];
    __shared__ float rv[1024];
    __shared__ int   ri[1024];
    __shared__ int   picked[10];
    const int tid = threadIdx.x;
    sv[tid] = vals[tid];
    __syncthreads();
    for (int it = 0; it < 10; ++it) {
        rv[tid] = sv[tid]; ri[tid] = tid;
        __syncthreads();
        for (int off = 512; off > 0; off >>= 1) {
            if (tid < off) {
                bool take = (rv[tid + off] > rv[tid]) ||
                            (rv[tid + off] == rv[tid] && ri[tid + off] < ri[tid]);
                if (take) { rv[tid] = rv[tid + off]; ri[tid] = ri[tid + off]; }
            }
            __syncthreads();
        }
        if (tid == 0) { picked[it] = ri[0]; sv[ri[0]] = -3.4e38f; }
        __syncthreads();
    }
    if (tid == 0) {
        for (int i = 1; i < 10; ++i) {
            int v = picked[i]; int j = i - 1;
            while (j >= 0 && picked[j] > v) { picked[j + 1] = picked[j]; --j; }
            picked[j + 1] = v;
        }
        for (int i = 0; i < 10; ++i) top[i] = picked[i];
    }
}

// ---------------------------------------------------------------------------
// Gather selected feature columns: seq[t][j] = h[t*1024 + top[j]]
// ---------------------------------------------------------------------------
__global__ void gather_kernel(const float* __restrict__ h,
                              const int* __restrict__ top,
                              float* __restrict__ seq, int n) {
    int i = blockIdx.x * blockDim.x + threadIdx.x;  // over 1024*10
    if (i < n) {
        int t = i / 10, j = i % 10;
        seq[i] = h[(size_t)t * 1024 + top[j]];
    }
}

// ---------------------------------------------------------------------------
// LSTM input projection (batched over t): gates[t][g] = x_t . Wih[g] + bih[g]+bhh[g]
// grid = T, block = 512 (one thread per gate row)
// ---------------------------------------------------------------------------
__global__ __launch_bounds__(512)
void lstm_in_kernel(const float* __restrict__ X, const float* __restrict__ Wih,
                    const float* __restrict__ bih, const float* __restrict__ bhh,
                    float* __restrict__ gates, int Kin) {
    const int t = blockIdx.x;
    const int g = threadIdx.x;
    const float* xr = X + (size_t)t * Kin;
    const float* wr = Wih + (size_t)g * Kin;
    float s = bih[g] + bhh[g];
    for (int k = 0; k < Kin; ++k) s += xr[k] * wr[k];
    gates[(size_t)t * 512 + g] = s;
}

// ---------------------------------------------------------------------------
// LSTM recurrence: one workgroup, Whh rows register-resident, h/c in LDS.
// gates = gates_in[t] + h @ Whh^T ; standard LSTM cell. Writes hs[t][0..127].
// ---------------------------------------------------------------------------
__global__ __launch_bounds__(512)
void lstm_rec_kernel(const float* __restrict__ gates_in,
                     const float* __restrict__ Whh,
                     float* __restrict__ hs, int T) {
    __shared__ float hsm[128];
    __shared__ float csm[128];
    __shared__ float gsm[512];
    const int tid = threadIdx.x;

    float w[128];
#pragma unroll
    for (int k = 0; k < 128; ++k) w[k] = Whh[(size_t)tid * 128 + k];

    if (tid < 128) { hsm[tid] = 0.0f; csm[tid] = 0.0f; }
    __syncthreads();

    for (int t = 0; t < T; ++t) {
        float gv = gates_in[(size_t)t * 512 + tid];
#pragma unroll
        for (int k = 0; k < 128; ++k) gv += w[k] * hsm[k];
        gsm[tid] = gv;
        __syncthreads();
        if (tid < 128) {
            float ig = sigm(gsm[tid]);
            float fg = sigm(gsm[128 + tid]);
            float gg = tanhf(gsm[256 + tid]);
            float og = sigm(gsm[384 + tid]);
            float c  = fg * csm[tid] + ig * gg;
            float hn = og * tanhf(c);
            csm[tid] = c;
            hsm[tid] = hn;
            hs[(size_t)t * 128 + tid] = hn;
        }
        __syncthreads();
    }
}

// ---------------------------------------------------------------------------
// Final FC: out[t] = hs[t] . fc3_W + fc3_b
// ---------------------------------------------------------------------------
__global__ void fc3_kernel(const float* __restrict__ hs, const float* __restrict__ W,
                           const float* __restrict__ b, float* __restrict__ out, int T) {
    int t = blockIdx.x * blockDim.x + threadIdx.x;
    if (t < T) {
        float s = b[0];
        const float* hr = hs + (size_t)t * 128;
#pragma unroll 8
        for (int k = 0; k < 128; ++k) s += hr[k] * W[k];
        out[t] = s;
    }
}

// ---------------------------------------------------------------------------
// Host orchestration
// ---------------------------------------------------------------------------
extern "C" void kernel_launch(void* const* d_in, const int* in_sizes, int n_in,
                              void* d_out, int out_size, void* d_ws, size_t ws_size,
                              hipStream_t stream) {
    constexpr int B = 1024, D = 1024, DFF = 2048, NLAYERS = 4;
    constexpr int HL = 128, GATES = 512, KTOP = 10;

    const float* x       = (const float*)d_in[0];
    const float* emb_W   = (const float*)d_in[1];
    const float* emb_b   = (const float*)d_in[2];
    const float* qkv_W   = (const float*)d_in[3];
    const float* qkv_b   = (const float*)d_in[4];
    const float* out_W   = (const float*)d_in[5];
    const float* out_b   = (const float*)d_in[6];
    const float* ff1_W   = (const float*)d_in[7];
    const float* ff1_b   = (const float*)d_in[8];
    const float* ff2_W   = (const float*)d_in[9];
    const float* ff2_b   = (const float*)d_in[10];
    const float* ln1_w   = (const float*)d_in[11];
    const float* ln1_b   = (const float*)d_in[12];
    const float* ln2_w   = (const float*)d_in[13];
    const float* ln2_b   = (const float*)d_in[14];
    const float* Wih0    = (const float*)d_in[15];
    const float* WihR    = (const float*)d_in[16];
    const float* Whh     = (const float*)d_in[17];
    const float* bih     = (const float*)d_in[18];
    const float* bhh     = (const float*)d_in[19];
    const float* fc3_W   = (const float*)d_in[20];
    const float* fc3_b   = (const float*)d_in[21];
    float* outp          = (float*)d_out;

    // workspace layout (256B aligned)
    char* ws = (char*)d_ws;
    unsigned short* hb16 = (unsigned short*)(ws + 0);          // 2 MB   B*D bf16
    float* hf32          = (float*)(ws + 2097152);             // 4 MB   B*D f32
    float* delta         = (float*)(ws + 6291456);             // 4 MB   B*D f32
    unsigned short* actb = (unsigned short*)(ws + 10485760);   // 4 MB   B*DFF bf16
    unsigned short* wslt = (unsigned short*)(ws + 14680064);   // 4 MB   weight slot bf16
    float* gates         = (float*)(ws + 18874368);            // 2 MB   B*512 f32
    float* hs0           = (float*)(ws + 20971520);            // 512 KB
    float* hs1           = (float*)(ws + 21495808);            // 512 KB
    float* seq           = (float*)(ws + 22020096);            // 40 KB
    float* colsums       = (float*)(ws + 22061056);            // 4 KB
    int*   topidx        = (int*)(ws + 22065152);              // 64 B
    (void)ws_size; (void)n_in; (void)in_sizes; (void)out_size;

    // --- embedding ---
    embed_kernel<<<(B * D) / 256, 256, 0, stream>>>(x, emb_W, emb_b, hf32, hb16);

    // --- encoder layers (S=1 => softmax is identity => o = v; skip Q,K) ---
    for (int l = 0; l < NLAYERS; ++l) {
        const float* Wv  = qkv_W + (size_t)l * 3 * D * D + (size_t)2 * D * D;
        const float* bv  = qkv_b + (size_t)l * 3 * D + 2 * D;
        const float* Wo  = out_W + (size_t)l * D * D;
        const float* bo  = out_b + (size_t)l * D;
        const float* W1  = ff1_W + (size_t)l * DFF * D;
        const float* b1  = ff1_b + (size_t)l * DFF;
        const float* W2  = ff2_W + (size_t)l * D * DFF;
        const float* b2  = ff2_b + (size_t)l * D;

        // v = h @ Wv^T + bv   (bf16 out only; consumed by next GEMM)
        cvt_f32_bf16<<<(D * D) / 256, 256, 0, stream>>>(Wv, wslt, D * D);
        gemm_bf16_wmma<<<dim3(B / 128, D / 64), 256, 0, stream>>>(
            hb16, wslt, bv, nullptr, actb, B, D, D, 0);

        // o = v @ Wo^T + bo   (f32 out for residual)
        cvt_f32_bf16<<<(D * D) / 256, 256, 0, stream>>>(Wo, wslt, D * D);
        gemm_bf16_wmma<<<dim3(B / 128, D / 64), 256, 0, stream>>>(
            actb, wslt, bo, delta, nullptr, B, D, D, 0);

        add_ln_kernel<<<B, 256, 0, stream>>>(hf32, delta, ln1_w + (size_t)l * D,
                                             ln1_b + (size_t)l * D, hb16, D);

        // f1 = relu(h @ W1^T + b1)   (bf16 out only)
        cvt_f32_bf16<<<(DFF * D) / 256, 256, 0, stream>>>(W1, wslt, DFF * D);
        gemm_bf16_wmma<<<dim3(B / 128, DFF / 64), 256, 0, stream>>>(
            hb16, wslt, b1, nullptr, actb, B, DFF, D, 1);

        // f2 = f1 @ W2^T + b2   (f32 out for residual)
        cvt_f32_bf16<<<(D * DFF) / 256, 256, 0, stream>>>(W2, wslt, D * DFF);
        gemm_bf16_wmma<<<dim3(B / 128, D / 64), 256, 0, stream>>>(
            actb, wslt, b2, delta, nullptr, B, D, DFF, 0);

        add_ln_kernel<<<B, 256, 0, stream>>>(hf32, delta, ln2_w + (size_t)l * D,
                                             ln2_b + (size_t)l * D, hb16, D);
    }

    // --- top-10 feature-column selection ---
    rowsum_kernel<<<B, 256, 0, stream>>>(hf32, colsums, D);
    topk_kernel<<<1, 1024, 0, stream>>>(colsums, topidx);
    gather_kernel<<<(B * KTOP + 255) / 256, 256, 0, stream>>>(hf32, topidx, seq, B * KTOP);

    // --- 3-layer LSTM (input projection batched; recurrence latency-optimized) ---
    lstm_in_kernel<<<B, GATES, 0, stream>>>(seq, Wih0, bih + 0, bhh + 0, gates, KTOP);
    lstm_rec_kernel<<<1, GATES, 0, stream>>>(gates, Whh + 0, hs0, B);

    lstm_in_kernel<<<B, GATES, 0, stream>>>(hs0, WihR + 0, bih + GATES, bhh + GATES,
                                            gates, HL);
    lstm_rec_kernel<<<1, GATES, 0, stream>>>(gates, Whh + (size_t)GATES * HL, hs1, B);

    lstm_in_kernel<<<B, GATES, 0, stream>>>(hs1, WihR + (size_t)GATES * HL,
                                            bih + 2 * GATES, bhh + 2 * GATES, gates, HL);
    lstm_rec_kernel<<<1, GATES, 0, stream>>>(gates, Whh + (size_t)2 * GATES * HL, hs0, B);

    // --- final FC ---
    fc3_kernel<<<(B + 255) / 256, 256, 0, stream>>>(hs0, fc3_W, fc3_b, outp, B);
}
